// MultiHeadAttentionAwareTemporalContex_q1d_k1d_773094114085
// MI455X (gfx1250) — matrix operand order repaired and checked by reading, not compiled, not run on hardware
//
#include <hip/hip_runtime.h>

#define BB  8
#define NNODE 307
#define TT  48
#define CC  256
#define HH  8
#define DKK 32
#define PADROWS 54   // 48 data rows + 6 zero rows for segment-boundary masking

typedef __bf16 bf16_t;
typedef __attribute__((ext_vector_type(4)))  __bf16 v4bf;
typedef __attribute__((ext_vector_type(8)))  __bf16 v8bf;
typedef __attribute__((ext_vector_type(16))) __bf16 v16bf;
typedef __attribute__((ext_vector_type(8)))  float  v8f;

__device__ __forceinline__ v8f wmma_bf16(v16bf a, v16bf b, v8f c) {
  // D = A(16x32 bf16) * B(32x16 bf16) + C(16x16 f32)
  return __builtin_amdgcn_wmma_f32_16x16x32_bf16(
      /*neg_a=*/false, a, /*neg_b=*/false, b,
      /*c_mod=*/(short)0, c, /*reuse_a=*/false, /*reuse_b=*/false);
}

__device__ __forceinline__ v16bf cat8(v8bf lo, v8bf hi) {
  return __builtin_shufflevector(lo, hi, 0,1,2,3,4,5,6,7,8,9,10,11,12,13,14,15);
}

// padded row index for time t: zero row before/after each segment (12,12,24)
__device__ __forceinline__ int padrow(int t) {
  return t + 1 + 2 * ((t >= 12) + (t >= 24));
}

// A fragment 16x32 bf16 from row-major [rows][ld].
// K order used here (contiguous k = 16h + s, same as fragB): lanes 0-15 row M=lane, K=0..15; lanes 16-31 same rows,
// K=16..31. Two 16-byte contiguous loads per lane.
__device__ __forceinline__ v16bf fragA(const bf16_t* m, int ld, int rbase, int kb) {
  int l = threadIdx.x & 31;
  const bf16_t* p = m + (rbase + (l & 15)) * ld + kb + ((l >> 4) << 4);
  return cat8(*(const v8bf*)p, *(const v8bf*)(p + 8));
}

// B fragment 32x16 bf16; source stored N-major [N][ldK] (K contiguous).
// K order used here (contiguous k = 16h + s, same as fragA): lanes 0-15 col N=lane, K=0..15; lanes 16-31 col N=lane-16, K=16..31.
__device__ __forceinline__ v16bf fragB(const bf16_t* m, int ld, int nbase, int kb) {
  int l = threadIdx.x & 31;
  const bf16_t* p = m + (nbase + (l & 15)) * ld + kb + ((l >> 4) << 4);
  return cat8(*(const v8bf*)p, *(const v8bf*)(p + 8));
}

// fp32 [48][256] global -> bf16 LDS in padded-segment layout [54][256]
__device__ __forceinline__ void stage(const float* __restrict__ g, bf16_t* xpad) {
  const float4* g4 = (const float4*)g;
  for (int i = threadIdx.x; i < TT * CC / 4; i += 256) {
    float4 f = g4[i];
    int t = i >> 6;            // (i*4)/256
    int c = (i & 63) << 2;
    v4bf p = { (bf16_t)f.x, (bf16_t)f.y, (bf16_t)f.z, (bf16_t)f.w };
    *(v4bf*)(xpad + padrow(t) * CC + c) = p;
  }
}

// Q/K conv projection: virtual A[t, dt*256+c] = x[t+dt-1, c] (segment masked via pad
// rows); [48 x 768] @ [768 x 256] -> dst [H][T][DK] bf16 in LDS.
__device__ __forceinline__ void conv_gemm(const bf16_t* xpad, const bf16_t* __restrict__ wT,
                                          const float* __restrict__ bias, bf16_t* dst) {
  int w = threadIdx.x >> 5;                       // wave owns cols [w*32, w*32+32) == head w
  int l = threadIdx.x & 31;
  // hoist per-lane row pointers (constant over the whole K loop)
  const bf16_t* rowp0 = xpad + padrow(0  + (l & 15)) * CC + ((l >> 4) << 4);
  const bf16_t* rowp1 = xpad + padrow(16 + (l & 15)) * CC + ((l >> 4) << 4);
  const bf16_t* rowp2 = xpad + padrow(32 + (l & 15)) * CC + ((l >> 4) << 4);
  const bf16_t* wp0 = wT + (w * 32      + (l & 15)) * 768 + ((l >> 4) << 4);
  const bf16_t* wp1 = wT + (w * 32 + 16 + (l & 15)) * 768 + ((l >> 4) << 4);

  v8f acc[3][2] = {};
  for (int ks = 0; ks < 24; ++ks) {
    int kb = ks * 32;
    int dt = kb >> 8;                             // 0,1,2 (constant within a K step)
    int off = (dt - 1) * CC + (kb & 255);         // shifted, pad rows supply zeros
    v16bf a0 = cat8(*(const v8bf*)(rowp0 + off), *(const v8bf*)(rowp0 + off + 8));
    v16bf a1 = cat8(*(const v8bf*)(rowp1 + off), *(const v8bf*)(rowp1 + off + 8));
    v16bf a2 = cat8(*(const v8bf*)(rowp2 + off), *(const v8bf*)(rowp2 + off + 8));
    v16bf b0 = cat8(*(const v8bf*)(wp0 + kb), *(const v8bf*)(wp0 + kb + 8));
    v16bf b1 = cat8(*(const v8bf*)(wp1 + kb), *(const v8bf*)(wp1 + kb + 8));
    acc[0][0] = wmma_bf16(a0, b0, acc[0][0]);
    acc[1][0] = wmma_bf16(a1, b0, acc[1][0]);
    acc[2][0] = wmma_bf16(a2, b0, acc[2][0]);
    acc[0][1] = wmma_bf16(a0, b1, acc[0][1]);
    acc[1][1] = wmma_bf16(a1, b1, acc[1][1]);
    acc[2][1] = wmma_bf16(a2, b1, acc[2][1]);
  }
  int col = l & 15, rtop = (l >> 4) << 3;
  for (int nt = 0; nt < 2; ++nt) {
    int dk = nt * 16 + col;
    float bs = bias[w * 32 + dk];
    for (int mt = 0; mt < 3; ++mt)
      for (int j = 0; j < 8; ++j) {
        int t = mt * 16 + rtop + j;
        dst[(w * TT + t) * DKK + dk] = (bf16_t)(acc[mt][nt][j] + bs);
      }
  }
}

// V linear: [48 x 256] @ [256 x 256] -> vT [H][DK][64] bf16 (t-contiguous, rows 48..63 zero)
__device__ __forceinline__ void v_gemm(const bf16_t* xpad, const bf16_t* __restrict__ wv,
                                       const float* __restrict__ bias, bf16_t* vt) {
  int w = threadIdx.x >> 5;
  int l = threadIdx.x & 31;
  const bf16_t* rowp0 = xpad + padrow(0  + (l & 15)) * CC + ((l >> 4) << 4);
  const bf16_t* rowp1 = xpad + padrow(16 + (l & 15)) * CC + ((l >> 4) << 4);
  const bf16_t* rowp2 = xpad + padrow(32 + (l & 15)) * CC + ((l >> 4) << 4);
  const bf16_t* wp0 = wv + (w * 32      + (l & 15)) * CC + ((l >> 4) << 4);
  const bf16_t* wp1 = wv + (w * 32 + 16 + (l & 15)) * CC + ((l >> 4) << 4);

  v8f acc[3][2] = {};
  for (int ks = 0; ks < 8; ++ks) {
    int kb = ks * 32;
    v16bf a0 = cat8(*(const v8bf*)(rowp0 + kb), *(const v8bf*)(rowp0 + kb + 8));
    v16bf a1 = cat8(*(const v8bf*)(rowp1 + kb), *(const v8bf*)(rowp1 + kb + 8));
    v16bf a2 = cat8(*(const v8bf*)(rowp2 + kb), *(const v8bf*)(rowp2 + kb + 8));
    v16bf b0 = cat8(*(const v8bf*)(wp0 + kb), *(const v8bf*)(wp0 + kb + 8));
    v16bf b1 = cat8(*(const v8bf*)(wp1 + kb), *(const v8bf*)(wp1 + kb + 8));
    acc[0][0] = wmma_bf16(a0, b0, acc[0][0]);
    acc[1][0] = wmma_bf16(a1, b0, acc[1][0]);
    acc[2][0] = wmma_bf16(a2, b0, acc[2][0]);
    acc[0][1] = wmma_bf16(a0, b1, acc[0][1]);
    acc[1][1] = wmma_bf16(a1, b1, acc[1][1]);
    acc[2][1] = wmma_bf16(a2, b1, acc[2][1]);
  }
  int col = l & 15, rtop = (l >> 4) << 3;
  for (int nt = 0; nt < 2; ++nt) {
    int dk = nt * 16 + col;
    float bs = bias[w * 32 + dk];
    for (int mt = 0; mt < 3; ++mt)
      for (int j = 0; j < 8; ++j) {
        int t = mt * 16 + rtop + j;
        vt[(w * DKK + dk) * 64 + t] = (bf16_t)(acc[mt][nt][j] + bs);
      }
  }
}

// Per-head attention: scores = Q Kt / sqrt(32), softmax over k-time, x = P V.
__device__ __forceinline__ void attention(const bf16_t* qbuf, const bf16_t* kbuf,
                                          const bf16_t* vt, bf16_t* pbuf, bf16_t* xbuf) {
  int h = threadIdx.x >> 5;                       // wave == head
  const bf16_t* qh = qbuf + h * TT * DKK;
  const bf16_t* kh = kbuf + h * TT * DKK;
  const bf16_t* vh = vt + h * DKK * 64;
  bf16_t* ph = pbuf + h * TT * 64;
  int l = threadIdx.x & 31, col = l & 15, rtop = (l >> 4) << 3;

  v8f s[3][3] = {};
  v16bf qa[3], kf[3];
  for (int mt = 0; mt < 3; ++mt) qa[mt] = fragA(qh, DKK, mt * 16, 0);
  for (int nt = 0; nt < 3; ++nt) kf[nt] = fragB(kh, DKK, nt * 16, 0);
  for (int mt = 0; mt < 3; ++mt)
    for (int nt = 0; nt < 3; ++nt)
      s[mt][nt] = wmma_bf16(qa[mt], kf[nt], s[mt][nt]);

  const float sc = 0.17677669529663687f;          // 1/sqrt(32)
  for (int mt = 0; mt < 3; ++mt) {
    for (int j = 0; j < 8; ++j) {
      // row (mt*16 + rtop + j) lives across 16 lanes x 3 N-tiles
      float m = fmaxf(fmaxf(s[mt][0][j], s[mt][1][j]), s[mt][2][j]);
      for (int d = 1; d < 16; d <<= 1) m = fmaxf(m, __shfl_xor(m, d, 16));
      float p0 = __expf((s[mt][0][j] - m) * sc);
      float p1 = __expf((s[mt][1][j] - m) * sc);
      float p2 = __expf((s[mt][2][j] - m) * sc);
      float sum = p0 + p1 + p2;
      for (int d = 1; d < 16; d <<= 1) sum += __shfl_xor(sum, d, 16);
      float inv = 1.0f / sum;
      int t = mt * 16 + rtop + j;
      ph[t * 64 +      col] = (bf16_t)(p0 * inv);
      ph[t * 64 + 16 + col] = (bf16_t)(p1 * inv);
      ph[t * 64 + 32 + col] = (bf16_t)(p2 * inv);
    }
  }

  v8f x[3][2] = {};
  for (int kk = 0; kk < 2; ++kk) {                // K over padded time 64
    int kb = kk * 32;
    v16bf a0 = fragA(ph, 64, 0,  kb);
    v16bf a1 = fragA(ph, 64, 16, kb);
    v16bf a2 = fragA(ph, 64, 32, kb);
    v16bf b0 = fragB(vh, 64, 0,  kb);
    v16bf b1 = fragB(vh, 64, 16, kb);
    x[0][0] = wmma_bf16(a0, b0, x[0][0]);
    x[1][0] = wmma_bf16(a1, b0, x[1][0]);
    x[2][0] = wmma_bf16(a2, b0, x[2][0]);
    x[0][1] = wmma_bf16(a0, b1, x[0][1]);
    x[1][1] = wmma_bf16(a1, b1, x[1][1]);
    x[2][1] = wmma_bf16(a2, b1, x[2][1]);
  }
  for (int nt = 0; nt < 2; ++nt) {
    int o = h * 32 + nt * 16 + col;
    for (int mt = 0; mt < 3; ++mt)
      for (int j = 0; j < 8; ++j) {
        int t = mt * 16 + rtop + j;
        xbuf[t * CC + o] = (bf16_t)(x[mt][nt][j]);   // plain [48][256] layout
      }
  }
}

__device__ __forceinline__ void out_gemm(const bf16_t* xbuf, const bf16_t* __restrict__ wo,
                                         const float* __restrict__ bo, float* __restrict__ gout) {
  int w = threadIdx.x >> 5;
  int l = threadIdx.x & 31;
  const bf16_t* wp0 = wo + (w * 32      + (l & 15)) * CC + ((l >> 4) << 4);
  const bf16_t* wp1 = wo + (w * 32 + 16 + (l & 15)) * CC + ((l >> 4) << 4);
  v8f acc[3][2] = {};
  for (int ks = 0; ks < 8; ++ks) {
    int kb = ks * 32;
    v16bf a0 = fragA(xbuf, CC, 0,  kb);
    v16bf a1 = fragA(xbuf, CC, 16, kb);
    v16bf a2 = fragA(xbuf, CC, 32, kb);
    v16bf b0 = cat8(*(const v8bf*)(wp0 + kb), *(const v8bf*)(wp0 + kb + 8));
    v16bf b1 = cat8(*(const v8bf*)(wp1 + kb), *(const v8bf*)(wp1 + kb + 8));
    acc[0][0] = wmma_bf16(a0, b0, acc[0][0]);
    acc[1][0] = wmma_bf16(a1, b0, acc[1][0]);
    acc[2][0] = wmma_bf16(a2, b0, acc[2][0]);
    acc[0][1] = wmma_bf16(a0, b1, acc[0][1]);
    acc[1][1] = wmma_bf16(a1, b1, acc[1][1]);
    acc[2][1] = wmma_bf16(a2, b1, acc[2][1]);
  }
  int col = l & 15, rtop = (l >> 4) << 3;
  for (int nt = 0; nt < 2; ++nt) {
    int o = w * 32 + nt * 16 + col;
    float bs = bo[o];
    for (int mt = 0; mt < 3; ++mt)
      for (int j = 0; j < 8; ++j) {
        int t = mt * 16 + rtop + j;
        gout[t * CC + o] = acc[mt][nt][j] + bs;
      }
  }
}

__global__ __launch_bounds__(256)
void fused_attn_kernel(const float* __restrict__ q_in, const float* __restrict__ k_in,
                       const float* __restrict__ v_in,
                       const bf16_t* __restrict__ wqT, const bf16_t* __restrict__ wkT,
                       const bf16_t* __restrict__ wv,  const bf16_t* __restrict__ wo,
                       const float* __restrict__ bq, const float* __restrict__ bk,
                       const float* __restrict__ bv, const float* __restrict__ bo,
                       float* __restrict__ out) {
  __shared__ __align__(16) bf16_t xpad[PADROWS * CC];     // 27.0 KB  padded stage / x
  __shared__ __align__(16) bf16_t qbuf[HH * TT * DKK];    // 24.0 KB  [h][t][dk]
  __shared__ __align__(16) bf16_t kbuf[HH * TT * DKK];    // 24.0 KB  [h][t][dk]
  __shared__ __align__(16) bf16_t vtbuf[HH * DKK * 64];   // 32.0 KB  [h][dk][t64]
  __shared__ __align__(16) bf16_t pbuf[HH * TT * 64];     // 48.0 KB  [h][t][t64]

  size_t base = (size_t)blockIdx.x * TT * CC;

  // Warm L2/WGP$ for the later-phase inputs (gfx1250 global_prefetch_b8 path).
  for (int i = threadIdx.x; i < (TT * CC) / 32; i += 256) {   // 128B cache lines
    __builtin_prefetch(k_in + base + i * 32, 0, 3);
    __builtin_prefetch(v_in + base + i * 32, 0, 3);
  }

  // zero-init padded regions (xpad zero rows, P cols 48..63, V rows 48..63)
  {
    uint4 zz = {0, 0, 0, 0};
    uint4* z0 = (uint4*)xpad;
    for (int i = threadIdx.x; i < (int)(sizeof(xpad) / 16); i += 256) z0[i] = zz;
    uint4* z1 = (uint4*)pbuf;
    for (int i = threadIdx.x; i < (int)(sizeof(pbuf) / 16); i += 256) z1[i] = zz;
    uint4* z2 = (uint4*)vtbuf;
    for (int i = threadIdx.x; i < (int)(sizeof(vtbuf) / 16); i += 256) z2[i] = zz;
  }
  __syncthreads();

  stage(q_in + base, xpad);
  __syncthreads();
  conv_gemm(xpad, wqT, bq, qbuf);
  __syncthreads();

  stage(k_in + base, xpad);
  __syncthreads();
  conv_gemm(xpad, wkT, bk, kbuf);
  __syncthreads();

  stage(v_in + base, xpad);
  __syncthreads();
  v_gemm(xpad, wv, bv, vtbuf);
  __syncthreads();

  attention(qbuf, kbuf, vtbuf, pbuf, xpad);
  __syncthreads();

  out_gemm(xpad, wo, bo, out + base);
}

// Prep: fp32 weights -> bf16, conv weights re-laid as [o][dt*256+c] (K-contiguous for B frags)
__global__ void prep_weights(const float* __restrict__ Wq, const float* __restrict__ Wk,
                             const float* __restrict__ Wv, const float* __restrict__ Wo,
                             bf16_t* __restrict__ wqT, bf16_t* __restrict__ wkT,
                             bf16_t* __restrict__ wv,  bf16_t* __restrict__ wo) {
  int idx = blockIdx.x * blockDim.x + threadIdx.x;
  const int NQ = 256 * 768;
  if (idx < NQ) {
    int o = idx / 768, r = idx % 768;
    int dt = r >> 8, c = r & 255;
    int src = (o * 256 + c) * 3 + dt;
    wqT[idx] = (bf16_t)Wq[src];
    wkT[idx] = (bf16_t)Wk[src];
  } else {
    int j = idx - NQ;
    if (j < 256 * 256) {
      wv[j] = (bf16_t)Wv[j];                      // Wv[o][c]: already N-major K-contiguous
      wo[j] = (bf16_t)Wo[j];
    }
  }
}

extern "C" void kernel_launch(void* const* d_in, const int* in_sizes, int n_in,
                              void* d_out, int out_size, void* d_ws, size_t ws_size,
                              hipStream_t stream) {
  const float* q_in = (const float*)d_in[0];
  const float* k_in = (const float*)d_in[1];
  const float* v_in = (const float*)d_in[2];
  const float* Wq = (const float*)d_in[3];
  const float* bq = (const float*)d_in[4];
  const float* Wk = (const float*)d_in[5];
  const float* bk = (const float*)d_in[6];
  const float* Wv = (const float*)d_in[7];
  const float* bv = (const float*)d_in[8];
  const float* Wo = (const float*)d_in[9];
  const float* bo = (const float*)d_in[10];

  bf16_t* wqT = (bf16_t*)d_ws;                    // [256][768]
  bf16_t* wkT = wqT + 256 * 768;                  // [256][768]
  bf16_t* wv  = wkT + 256 * 768;                  // [256][256]
  bf16_t* wo  = wv  + 256 * 256;                  // [256][256]  (total 1 MB of ws)

  const int prep_threads = 256 * 768 + 256 * 256; // 262144
  prep_weights<<<prep_threads / 256, 256, 0, stream>>>(Wq, Wk, Wv, Wo, wqT, wkT, wv, wo);

  fused_attn_kernel<<<BB * NNODE, 256, 0, stream>>>(
      q_in, k_in, v_in, wqT, wkT, wv, wo, bq, bk, bv, bo, (float*)d_out);
}
